// Visual_Attention_206158430434
// MI455X (gfx1250) — matrix operand order repaired and checked by reading, not compiled, not run on hardware
//
#include <hip/hip_runtime.h>
#include <hip/hip_bf16.h>

typedef __attribute__((ext_vector_type(16))) __bf16 v16bf;
typedef __attribute__((ext_vector_type(8)))  float  v8f;
typedef __attribute__((ext_vector_type(4)))  unsigned int u32x4;
typedef __attribute__((ext_vector_type(8)))  int  i32x8;
typedef __attribute__((ext_vector_type(4)))  int  i32x4;

#if defined(__HIP_DEVICE_COMPILE__) && defined(__has_builtin)
#if __has_builtin(__builtin_amdgcn_tensor_load_to_lds)
#define HAVE_TDM 1
#endif
#endif

// ---------------------------------------------------------------------------
// WMMA helpers (bf16 16x16x32, f32 accumulate)
// ---------------------------------------------------------------------------
__device__ inline v8f wmma_bf16(v16bf a, v16bf b, v8f c) {
    return __builtin_amdgcn_wmma_f32_16x16x32_bf16(
        false, a, false, b, (short)0, c, false, false);
}

// A fragment: 16x32 tile from row-major LDS.  Lane<16: row m=lane, K runs
// {0..15}; lane>=16: K runs {16..31}.  Two 16B LDS loads.
__device__ inline v16bf load_a_frag(const __bf16* base, int pitch,
                                    int m_base, int k_base, int lane) {
    int half = lane >> 4;
    int m    = lane & 15;
    const __bf16* row = base + (size_t)(m_base + m) * pitch + k_base;
    v16bf a;
    __builtin_memcpy(&a,                               row + half * 16,     16);
    __builtin_memcpy(reinterpret_cast<char*>(&a) + 16, row + half * 16 + 8, 16);
    return a;
}

// B fragment: 32x16 tile from N-major LDS (Bs[n][k]).  Lane n holds col n,
// 16 contiguous K values (K offset 16 for upper half-wave).  One 32B load.
__device__ inline v16bf load_b_frag(const __bf16* base, int pitch,
                                    int n_base, int k_base, int lane) {
    int half = lane >> 4;
    int n    = lane & 15;
    const __bf16* row = base + (size_t)(n_base + n) * pitch + k_base + half * 16;
    v16bf b;
    __builtin_memcpy(&b, row, 32);
    return b;
}

// ---------------------------------------------------------------------------
// TDM: 2D tile load, bf16, row length 512 elems, LDS pitch 520 elems
// (pad 4 DWORDs after every 256 DWORDs -> 1024B row + 16B pad = 1040B pitch).
// ---------------------------------------------------------------------------
#ifdef HAVE_TDM
__device__ inline void tdm_load_rows(const __bf16* gptr, __bf16* lptr, int rows) {
    unsigned long long ga = (unsigned long long)(uintptr_t)gptr;
    unsigned int lds      = (unsigned int)(uintptr_t)lptr;   // generic low32 = LDS offset
    u32x4 g0;
    g0[0] = 1u;                                        // count=1, user mode
    g0[1] = lds;                                       // lds_addr
    g0[2] = (unsigned int)(ga & 0xFFFFFFFFu);          // global_addr[31:0]
    g0[3] = (unsigned int)((ga >> 32) & 0x01FFFFFFu)   // global_addr[56:32]
          | (2u << 30);                                // type = 2 (image)
    i32x8 g1;
    g1[0] = (1 << 16)        // data_size = 1 (2 bytes)
          | (1 << 20)        // pad_enable
          | (7 << 22)        // pad_interval: 256 DWORDs (= 1024B row)
          | (3 << 25);       // pad_amount: 4 DWORDs (= 16B)
    g1[1] = (512 << 16);     // tensor_dim0 = 512 (bits 48..63)
    g1[2] = (8192 << 16);    // tensor_dim1 low16 = 8192 (bits 80..95)
    g1[3] = (512 << 16);     // tensor_dim1 hi = 0 | tile_dim0 = 512 (bits 112..127)
    g1[4] = rows;            // tile_dim1 (bits 128..143), tile_dim2 = 0
    g1[5] = 512;             // tensor_dim0_stride low32 = 512 elems
    g1[6] = 0;
    g1[7] = 0;
    i32x4 z4 = {0, 0, 0, 0};
    i32x8 z8 = {0, 0, 0, 0, 0, 0, 0, 0};
    __builtin_amdgcn_tensor_load_to_lds(g0, g1, z4, z4, z8, 0);
}
#endif

__device__ inline void wait_tensor0() {
#if defined(__HIP_DEVICE_COMPILE__) && defined(__has_builtin)
#if __has_builtin(__builtin_amdgcn_s_wait_tensorcnt)
    __builtin_amdgcn_s_wait_tensorcnt(0);
#else
    asm volatile("s_wait_tensorcnt 0x0" ::: "memory");
#endif
#endif
}

// ---------------------------------------------------------------------------
// Kernel 1: projection  Y[M,512](bf16) = X[M,1024](f32) @ W[1024,512] + b
// Tile: 128(M) x 64(N) per 256-thread block (8 waves), BK=32.
// ---------------------------------------------------------------------------
__global__ __launch_bounds__(256)
void proj_kernel(const float* __restrict__ X, const float* __restrict__ W,
                 const float* __restrict__ Bv, __bf16* __restrict__ Y) {
    const int K = 1024, N = 512;
    __shared__ __bf16 Xs[128][40];
    __shared__ __bf16 Ws[64][40];   // N-major: Ws[n][k]

    const int t    = threadIdx.x;
    const int lane = t & 31;
    const int wave = t >> 5;
    const int mBase = blockIdx.x * 128;
    const int nBase = blockIdx.y * 64;

    v8f acc[4] = {};

    for (int k0 = 0; k0 < K; k0 += 32) {
        for (int i = 0; i < 4; ++i) {
            int e = (t + i * 256) * 4;
            int r = e >> 5, c = e & 31;
            float f[4];
            __builtin_memcpy(f, X + (size_t)(mBase + r) * K + k0 + c, 16);
            __bf16* d = &Xs[r][c];
            d[0] = (__bf16)f[0]; d[1] = (__bf16)f[1];
            d[2] = (__bf16)f[2]; d[3] = (__bf16)f[3];
        }
        for (int i = 0; i < 8; ++i) {
            int e  = t + i * 256;
            int kk = e >> 6, n = e & 63;
            Ws[n][kk] = (__bf16)W[(size_t)(k0 + kk) * N + nBase + n];
        }
        __syncthreads();

        v16bf a = load_a_frag(&Xs[0][0], 40, wave * 16, 0, lane);
        for (int c = 0; c < 4; ++c) {
            v16bf b = load_b_frag(&Ws[0][0], 40, c * 16, 0, lane);
            acc[c] = wmma_bf16(a, b, acc[c]);
        }
        __syncthreads();
    }

    const int half = lane >> 4, nl = lane & 15;
    for (int c = 0; c < 4; ++c) {
        int col = nBase + c * 16 + nl;
        float bias = Bv[col];
        for (int r = 0; r < 8; ++r) {
            int row = mBase + wave * 16 + r + half * 8;
            Y[(size_t)row * N + col] = (__bf16)(acc[c][r] + bias);
        }
    }
}

// ---------------------------------------------------------------------------
// Kernel 2: flash attention.  out[N,512](f32) = softmax(qp kp^T / 64) vp
// 32 query rows/block, key tiles of 64, d=512 in register accumulators.
// LDS layout (bytes):
//   Qs [32][520] bf16 @ 0        (33280)
//   Ks [64][520] bf16 @ 33280    (66560)   TDM target, prefetched
//   Vt [512][72] bf16 @ 99840    (73728)   software-transposed V
//   Sf [32][68]  f32  @ 173568   ( 8704)
//   Ps [32][72]  bf16 @ 182272   ( 4608)
//   m/l/scale [3][32] @ 186880   (  384)   total 187264
// ---------------------------------------------------------------------------
#define ATTN_SMEM_BYTES 187264

__global__ __launch_bounds__(256)
void attn_kernel(const __bf16* __restrict__ qp, const __bf16* __restrict__ kp,
                 const __bf16* __restrict__ vp, float* __restrict__ out) {
    extern __shared__ char smem[];
    const int D = 512;
    __bf16* Qs  = (__bf16*)(smem);
    __bf16* Ks  = (__bf16*)(smem + 33280);
    __bf16* Vt  = (__bf16*)(smem + 99840);
    float*  Sf  = (float*)(smem + 173568);
    __bf16* Ps  = (__bf16*)(smem + 182272);
    float*  mrow = (float*)(smem + 186880);
    float*  lrow = mrow + 32;
    float*  scl  = mrow + 64;

    const int t    = threadIdx.x;
    const int lane = t & 31;
    const int wave = t >> 5;
    const int half = lane >> 4;
    const int nl   = lane & 15;
    const int qBase = blockIdx.x * 32;

#ifdef HAVE_TDM
    if (wave == 0) {
        tdm_load_rows(qp + (size_t)qBase * D, Qs, 32);   // Q tile once
        tdm_load_rows(kp, Ks, 64);                       // K tile 0
    }
#else
    for (int i = 0; i < 8; ++i) {
        int e = (t + i * 256) * 8;
        int r = e >> 9, c = e & 511;
        __builtin_memcpy(&Qs[r * 520 + c], qp + (size_t)(qBase + r) * D + c, 16);
    }
#endif
    if (t < 32) { mrow[t] = -3.0e38f; lrow[t] = 0.0f; }

    v8f acc[2][4] = {};   // [row-tile][col-tile of wave's 64 d-cols]

    for (int kt = 0; kt < 128; ++kt) {
        const int kBase = kt * 64;
#ifdef HAVE_TDM
        if (wave == 0) wait_tensor0();   // K(t) (and Q on t=0) landed
        __syncthreads();
#else
        __syncthreads();
        for (int i = 0; i < 16; ++i) {
            int e = (t + i * 256) * 8;
            int r = e >> 9, c = e & 511;
            __builtin_memcpy(&Ks[r * 520 + c], kp + (size_t)(kBase + r) * D + c, 16);
        }
        __syncthreads();
#endif

        // S = Qs @ Ks^T : each wave one 16x16 tile of [32 x 64]
        const int rw = wave >> 2, cw = wave & 3;
        v8f s = {};
        for (int d0 = 0; d0 < 512; d0 += 32) {
            v16bf a = load_a_frag(Qs, 520, rw * 16, d0, lane);
            v16bf b = load_b_frag(Ks, 520, cw * 16, d0, lane);
            s = wmma_bf16(a, b, s);
        }
        for (int r = 0; r < 8; ++r)
            Sf[(rw * 16 + r + half * 8) * 68 + cw * 16 + nl] = s[r] * (1.0f / 64.0f);
        __syncthreads();                  // all waves finished reading Ks

#ifdef HAVE_TDM
        if (wave == 0 && kt + 1 < 128)    // prefetch K(t+1); overlaps softmax+PV
            tdm_load_rows(kp + (size_t)(kBase + 64) * D, Ks, 64);
#endif

        // Online softmax: one thread per query row
        if (t < 32) {
            float mo = mrow[t];
            float mx = mo;
            for (int c = 0; c < 64; ++c) mx = fmaxf(mx, Sf[t * 68 + c]);
            float sum = 0.0f;
            for (int c = 0; c < 64; ++c) {
                float p = __expf(Sf[t * 68 + c] - mx);
                sum += p;
                Ps[t * 72 + c] = (__bf16)p;
            }
            float sc = __expf(mo - mx);
            scl[t]  = sc;
            mrow[t] = mx;
            lrow[t] = lrow[t] * sc + sum;
        }
        __syncthreads();

        // Rescale accumulators by exp(m_old - m_new)
        for (int rt = 0; rt < 2; ++rt)
            for (int r = 0; r < 8; ++r) {
                float f = scl[rt * 16 + r + half * 8];
                for (int ct = 0; ct < 4; ++ct) acc[rt][ct][r] *= f;
            }

        // V tile -> Vt[d][key] transposed; 2-key x 8-d blocks, paired b32 stores
        for (int i = 0; i < 8; ++i) {
            int b    = t + i * 256;          // 0..2047
            int key2 = (b >> 6) << 1;        // even key
            int d8   = (b & 63) << 3;
            __bf16 lo[8], hi[8];
            __builtin_memcpy(lo, vp + (size_t)(kBase + key2) * D + d8, 16);
            __builtin_memcpy(hi, vp + (size_t)(kBase + key2 + 1) * D + d8, 16);
            for (int j = 0; j < 8; ++j) {
                __bf16 pair[2] = {lo[j], hi[j]};
                __builtin_memcpy(&Vt[(d8 + j) * 72 + key2], pair, 4);
            }
        }
        __syncthreads();

        // O += P @ V : wave owns d-cols [64*wave, 64*wave+63]
        for (int kk = 0; kk < 64; kk += 32) {
            for (int rt = 0; rt < 2; ++rt) {
                v16bf a = load_a_frag(Ps, 72, rt * 16, kk, lane);
                for (int ct = 0; ct < 4; ++ct) {
                    v16bf b = load_b_frag(Vt, 72, wave * 64 + ct * 16, kk, lane);
                    acc[rt][ct] = wmma_bf16(a, b, acc[rt][ct]);
                }
            }
        }
    }
    __syncthreads();

    // Epilogue: divide by running row sum, store f32
    for (int rt = 0; rt < 2; ++rt) {
        for (int r = 0; r < 8; ++r) {
            int lr  = rt * 16 + r + half * 8;
            float inv = 1.0f / lrow[lr];
            int row = qBase + lr;
            for (int ct = 0; ct < 4; ++ct) {
                int col = wave * 64 + ct * 16 + nl;
                out[(size_t)row * 512 + col] = acc[rt][ct][r] * inv;
            }
        }
    }
}

// ---------------------------------------------------------------------------
// Launch
// ---------------------------------------------------------------------------
extern "C" void kernel_launch(void* const* d_in, const int* in_sizes, int n_in,
                              void* d_out, int out_size, void* d_ws, size_t ws_size,
                              hipStream_t stream) {
    const float* q  = (const float*)d_in[0];
    const float* k  = (const float*)d_in[1];
    const float* v  = (const float*)d_in[2];
    const float* Wq = (const float*)d_in[3];
    const float* bq = (const float*)d_in[4];
    const float* Wk = (const float*)d_in[5];
    const float* bk = (const float*)d_in[6];
    const float* Wv = (const float*)d_in[7];
    const float* bv = (const float*)d_in[8];

    const size_t NPROJ = (size_t)8192 * 512;
    __bf16* qp = (__bf16*)d_ws;
    __bf16* kp = qp + NPROJ;
    __bf16* vp = kp + NPROJ;

    dim3 pgrid(64, 8);
    dim3 pblock(256);
    proj_kernel<<<pgrid, pblock, 0, stream>>>(q, Wq, bq, qp);
    proj_kernel<<<pgrid, pblock, 0, stream>>>(k, Wk, bk, kp);
    proj_kernel<<<pgrid, pblock, 0, stream>>>(v, Wv, bv, vp);

    attn_kernel<<<dim3(256), dim3(256), ATTN_SMEM_BYTES, stream>>>(
        qp, kp, vp, (float*)d_out);
}
